// Gate_GraphTransformer_24197845746277
// MI455X (gfx1250) — matrix operand order repaired and checked by reading, not compiled, not run on hardware
//
#include <hip/hip_runtime.h>
#include <hip/hip_bf16.h>
#include <math.h>

// ---------------------------------------------------------------------------
// Gate_GraphTransformer on MI455X (gfx1250, wave32, WMMA bf16, f32 accum).
// All operands pre-packed to zero-padded bf16 (1536x1536, heads padded to 512)
// so the GEMM hot loop has no conversions and no bounds checks:
//   staging  : 4x global_load_b128 + 4x ds_store_b128 per thread per K-tile
//   fragments: 2x ds_load_b128 per 16x16 fragment (fragment-order LDS layout)
//   compute  : 8x v_wmma_f32_16x16x32_bf16 per wave per K-step (32x64 patch)
// ---------------------------------------------------------------------------

#define NREAL 1527
#define NPAD  1536
#define DK    509
#define DKP   512            // per-head padded K (3*512 == 1536)

typedef unsigned short u16;
typedef __attribute__((ext_vector_type(16))) __bf16          v16bf;
typedef __attribute__((ext_vector_type(16))) unsigned short  v16u;
typedef __attribute__((ext_vector_type(8)))  float           v8f;

union Frag { v16u v; uint4 q[2]; };

// round-to-nearest-even f32 -> bf16 bits
__device__ __forceinline__ u16 f2bf(float f) {
    unsigned u = __builtin_bit_cast(unsigned, f);
    u += 0x7FFFu + ((u >> 16) & 1u);
    return (u16)(u >> 16);
}

// ---------------------------------------------------------------------------
// NT GEMM on packed bf16:  C[m][n] = alpha * sum_k A[m][kOffA+k]*Bt[n][kOffB+k]
//                                    + beta*Cadd + bias,  zero-filled pads.
// A: [*, lda] bf16, Bt: [*, ldb] bf16 (B already transposed at pack time).
// Outputs fp32 Cf and/or bf16 Cb (optionally head-padded column mapping).
// Grid: (gridN, 12) tiles of 128x128; 256 threads = 8 waves of 32x64.
// ---------------------------------------------------------------------------
__global__ __launch_bounds__(256)
void gemm_bf16_nt(const u16* __restrict__ A, int lda, int kOffA,
                  const u16* __restrict__ Bt, int ldb, int kOffB,
                  int Kt, float alpha,
                  float* __restrict__ Cf, const float* __restrict__ Cadd, float beta,
                  u16* __restrict__ Cb, int ldc,
                  int Mreal, int Nreal,
                  const float* __restrict__ bias, int biasMode,  // 0 none,1 col,2 row
                  int headPadCb)
{
    __shared__ u16 As[128][40];   // [m][k], 80B row stride: 16B-aligned runs
    __shared__ u16 Bs[128][40];   // [n][k]

    const int tid   = threadIdx.x;
    const int lane  = tid & 31;
    const int wave  = tid >> 5;            // 0..7
    const int wm    = (wave & 3) * 32;
    const int wn    = (wave >> 2) * 64;
    const int m0    = blockIdx.y * 128;
    const int n0    = blockIdx.x * 128;
    const int lhalf = lane >> 4;           // 0/1
    const int l16   = lane & 15;
    const int sr    = tid >> 1;            // staging row 0..127
    const int sseg  = (tid & 1) << 4;      // 0 or 16 (u16 elements)

    const v8f vz = {0.f,0.f,0.f,0.f,0.f,0.f,0.f,0.f};
    v8f acc[2][4];
    #pragma unroll
    for (int i = 0; i < 2; ++i)
        #pragma unroll
        for (int j = 0; j < 4; ++j) acc[i][j] = vz;

    const u16* aBase = A  + (size_t)(m0 + sr) * lda + kOffA + sseg;
    const u16* bBase = Bt + (size_t)(n0 + sr) * ldb + kOffB + sseg;

    for (int k0 = 0; k0 < Kt; k0 += 32) {
        // ---- stage: 2x b128 loads + 2x b128 LDS stores per operand ----
        const uint4* pa = (const uint4*)(aBase + k0);
        const uint4* pb = (const uint4*)(bBase + k0);
        uint4 a0 = pa[0], a1 = pa[1];
        uint4 b0 = pb[0], b1 = pb[1];
        if (k0 + 32 < Kt) {          // global_prefetch next K tile
            __builtin_prefetch((const char*)pa + 64, 0, 1);
            __builtin_prefetch((const char*)pb + 64, 0, 1);
        }
        *(uint4*)&As[sr][sseg]     = a0;
        *(uint4*)&As[sr][sseg + 8] = a1;
        *(uint4*)&Bs[sr][sseg]     = b0;
        *(uint4*)&Bs[sr][sseg + 8] = b1;
        __syncthreads();

        // ---- fragments: contiguous 16B runs per documented wave32 layouts ----
        Frag fa[2], fb[4];
        #pragma unroll
        for (int fm = 0; fm < 2; ++fm) {
            int row = wm + fm * 16 + l16;
            // A 16x32 bf16: lane(hi) holds K = hi*8..hi*8+7 and 16+hi*8..+7
            fa[fm].q[0] = *(const uint4*)&As[row][lhalf * 8];
            fa[fm].q[1] = *(const uint4*)&As[row][16 + lhalf * 8];
        }
        #pragma unroll
        for (int fn = 0; fn < 4; ++fn) {
            int col = wn + fn * 16 + l16;
            // B 32x16 bf16: lane(hi) holds K = hi*8..hi*8+7 and 16+hi*8..+7
            fb[fn].q[0] = *(const uint4*)&Bs[col][lhalf * 8];
            fb[fn].q[1] = *(const uint4*)&Bs[col][16 + lhalf * 8];
        }

        #pragma unroll
        for (int fm = 0; fm < 2; ++fm) {
            v16bf av = __builtin_bit_cast(v16bf, fa[fm].v);
            #pragma unroll
            for (int fn = 0; fn < 4; ++fn) {
                v16bf bv = __builtin_bit_cast(v16bf, fb[fn].v);
                acc[fm][fn] = __builtin_amdgcn_wmma_f32_16x16x32_bf16(
                    false, av, false, bv, (short)0, acc[fm][fn], false, false);
            }
        }
        __syncthreads();
    }

    // ---- epilogue: C/D layout row = 8*lhalf + r, col = l16; zero-fill pads ----
    #pragma unroll
    for (int fm = 0; fm < 2; ++fm)
        #pragma unroll
        for (int fn = 0; fn < 4; ++fn) {
            int col = n0 + wn + fn * 16 + l16;
            #pragma unroll
            for (int r = 0; r < 8; ++r) {
                int row = m0 + wm + fm * 16 + lhalf * 8 + r;
                float v = 0.f;
                if (row < Mreal && col < Nreal) {
                    v = alpha * acc[fm][fn][r];
                    if (Cadd) v += beta * Cadd[(size_t)row * ldc + col];
                    if (bias) v += (biasMode == 2) ? bias[row] : bias[col];
                }
                if (Cf) Cf[(size_t)row * ldc + col] = v;
                if (Cb) {
                    int c2 = col;
                    if (headPadCb) {   // bijective map: 3 heads of 509 -> 512
                        if (col < NREAL) c2 = (col / DK) * DKP + (col % DK);
                        else { int g = col - NREAL; c2 = (g / 3) * DKP + DK + (g % 3); }
                    }
                    Cb[(size_t)row * ldc + c2] = f2bf(v);
                }
            }
        }
}

// ---------------------------------------------------------------------------
// fp32 [rows x cols] (row stride srcStride) -> zero-padded bf16 [1536 x 1536]
// ---------------------------------------------------------------------------
__global__ __launch_bounds__(256)
void pack_bf16_kernel(const float* __restrict__ src, int srcStride,
                      int rows, int cols, u16* __restrict__ dst)
{
    size_t idx = (size_t)blockIdx.x * 256 + threadIdx.x;
    size_t NP2 = (size_t)NPAD * NPAD;
    if (idx >= NP2) return;
    int i = (int)(idx / NPAD), j = (int)(idx % NPAD);
    float v = (i < rows && j < cols) ? src[(size_t)i * srcStride + j] : 0.f;
    dst[idx] = f2bf(v);
}

// row sums of feature with zero diagonal
__global__ __launch_bounds__(256)
void rowsum_kernel(const float* __restrict__ A, float* __restrict__ rs)
{
    __shared__ float red[256];
    int row = blockIdx.x;
    float s = 0.f;
    for (int j = threadIdx.x; j < NREAL; j += 256)
        if (j != row) s += A[(size_t)row * NREAL + j];
    red[threadIdx.x] = s; __syncthreads();
    for (int st = 128; st > 0; st >>= 1) {
        if (threadIdx.x < st) red[threadIdx.x] += red[threadIdx.x + st];
        __syncthreads();
    }
    if (threadIdx.x == 0) rs[row] = red[0];
}

// PTb = P^T bf16 ; U0b = E^T bf16 ; ETf = E^T f32 ; rg = eta0*E ;
// Fb = E bf16 ; Hf0 = E f32     (all zero-padded to 1536x1536)
__global__ __launch_bounds__(256)
void rwr_init_kernel(const float* __restrict__ feat, const float* __restrict__ rs,
                     const float* __restrict__ etas,
                     u16* __restrict__ PTb, u16* __restrict__ U0b,
                     float* __restrict__ ETf, float* __restrict__ rg,
                     u16* __restrict__ Fb, float* __restrict__ Hf0)
{
    size_t idx = (size_t)blockIdx.x * 256 + threadIdx.x;
    size_t NP2 = (size_t)NPAD * NPAD;
    if (idx >= NP2) return;
    int i = (int)(idx / NPAD), j = (int)(idx % NPAD);
    bool in = (i < NREAL) && (j < NREAL);
    float e  = in ? feat[(size_t)i * NREAL + j] : 0.f;  // E[i][j]
    float et = in ? feat[(size_t)j * NREAL + i] : 0.f;  // E^T[i][j]
    float pt = 0.f;
    if (in && i != j) {
        float s = rs[j];
        pt = feat[(size_t)j * NREAL + i] / (s == 0.f ? 1.f : s);  // P[j][i]
    }
    PTb[idx] = f2bf(pt);
    U0b[idx] = f2bf(et);
    ETf[idx] = et;
    rg[idx]  = etas[0] * e;
    Fb[idx]  = f2bf(e);
    Hf0[idx] = e;
}

// rg += etas[e] * U   (U == W^T already, elementwise)
__global__ __launch_bounds__(256)
void rg_acc_kernel(float* __restrict__ rg, const float* __restrict__ Uf,
                   const float* __restrict__ etas, int e)
{
    size_t idx = (size_t)blockIdx.x * 256 + threadIdx.x;
    size_t NP2 = (size_t)NPAD * NPAD;
    if (idx >= NP2) return;
    rg[idx] += etas[e] * Uf[idx];
}

// one block per padded row: attn = softmax(S*rg) -> bf16, zero pads
__global__ __launch_bounds__(256)
void bias_softmax_bf16(const float* __restrict__ S, const float* __restrict__ rg,
                       u16* __restrict__ Sb)
{
    __shared__ float red[256];
    int row = blockIdx.x, tid = threadIdx.x;
    u16* orow = Sb + (size_t)row * NPAD;
    if (row >= NREAL) {
        for (int j = tid; j < NPAD; j += 256) orow[j] = 0;
        return;
    }
    const float* srow = S  + (size_t)row * NPAD;
    const float* grow = rg + (size_t)row * NPAD;
    float vals[6];
    float mx = -3.4e38f;
    #pragma unroll
    for (int k = 0; k < 6; ++k) {
        int j = tid + k * 256;
        float v = (j < NREAL) ? srow[j] * grow[j] : -3.4e38f;
        vals[k] = v; mx = fmaxf(mx, v);
    }
    red[tid] = mx; __syncthreads();
    for (int st = 128; st > 0; st >>= 1) {
        if (tid < st) red[tid] = fmaxf(red[tid], red[tid + st]);
        __syncthreads();
    }
    mx = red[0]; __syncthreads();
    float sum = 0.f;
    #pragma unroll
    for (int k = 0; k < 6; ++k) {
        int j = tid + k * 256;
        float e = (j < NREAL) ? __expf(vals[k] - mx) : 0.f;
        vals[k] = e; sum += e;
    }
    red[tid] = sum; __syncthreads();
    for (int st = 128; st > 0; st >>= 1) {
        if (tid < st) red[tid] += red[tid + st];
        __syncthreads();
    }
    float inv = 1.f / red[0];
    #pragma unroll
    for (int k = 0; k < 6; ++k) {
        int j = tid + k * 256;
        orow[j] = (j < NREAL) ? f2bf(vals[k] * inv) : (u16)0;
    }
}

// zero bf16 columns [c0, NPAD) for all rows
__global__ __launch_bounds__(256)
void zero_cols_kernel(u16* __restrict__ buf, int c0)
{
    int r = blockIdx.x * 256 + threadIdx.x;
    if (r >= NPAD) return;
    for (int c = c0; c < NPAD; ++c) buf[(size_t)r * NPAD + c] = 0;
}

// out = (1-sigmoid(G+b))*feature + sigmoid(G+b)*faeture  (unpadded 1527x1527)
__global__ __launch_bounds__(256)
void gate_out_kernel(const float* __restrict__ G, const float* __restrict__ gate_b,
                     const float* __restrict__ feature, const float* __restrict__ faeture,
                     float* __restrict__ out)
{
    size_t idx = (size_t)blockIdx.x * 256 + threadIdx.x;
    size_t total = (size_t)NREAL * NREAL;
    if (idx >= total) return;
    int i = (int)(idx / NREAL), j = (int)(idx % NREAL);
    float g = 1.f / (1.f + __expf(-(G[(size_t)i * NPAD + j] + gate_b[j])));
    out[idx] = (1.f - g) * feature[idx] + g * faeture[idx];
}

// ---------------------------------------------------------------------------
extern "C" void kernel_launch(void* const* d_in, const int* in_sizes, int n_in,
                              void* d_out, int out_size, void* d_ws, size_t ws_size,
                              hipStream_t stream)
{
    (void)in_sizes; (void)n_in; (void)out_size; (void)ws_size;

    const float* feature = (const float*)d_in[0];
    const float* faeture = (const float*)d_in[1];
    const float* Wq      = (const float*)d_in[2];
    const float* bq      = (const float*)d_in[3];
    const float* Wk      = (const float*)d_in[4];
    const float* bk      = (const float*)d_in[5];
    const float* Wv      = (const float*)d_in[6];
    const float* bv      = (const float*)d_in[7];
    const float* Wo      = (const float*)d_in[8];
    const float* bo      = (const float*)d_in[9];
    const float* gate_W  = (const float*)d_in[10];
    const float* gate_b  = (const float*)d_in[11];
    const float* etas    = (const float*)d_in[12];
    float* out = (float*)d_out;

    const size_t NP2 = (size_t)NPAD * NPAD;
    const size_t D2  = (size_t)NREAL * NREAL;

    // ---- workspace carve: fp32 first, then bf16 (all NP2-sized, aligned) ----
    float* f  = (float*)d_ws;
    float* rs  = f;  f += NPAD;
    float* ETf = f;  f += NP2;
    float* rg  = f;  f += NP2;
    float* Uf  = f;  f += NP2;
    float* Hf0 = f;  f += NP2;
    float* Hf1 = f;  f += NP2;
    float* Sfp = f;  f += NP2;
    float* Gf  = f;  f += NP2;
    u16* u = (u16*)f;
    u16* PTb  = u;  u += NP2;
    u16* Ub0  = u;  u += NP2;
    u16* Ub1  = u;  u += NP2;
    u16* Fb   = u;  u += NP2;
    u16* Hb1  = u;  u += NP2;
    u16* Hb2  = u;  u += NP2;
    u16* Qb   = u;  u += NP2;
    u16* Kb   = u;  u += NP2;
    u16* Vtb  = u;  u += NP2;
    u16* Sbb  = u;  u += NP2;
    u16* CTXb = u;  u += NP2;
    u16* Wb   = u;  u += NP2;   // reusable packed-weight buffer

    const int egrid = (int)((NP2 + 255) / 256);

    auto gemm = [&](const u16* A, int kOffA, const u16* Bt, int kOffB, int Kt,
                    float alpha, float* Cf, const float* Cadd, float beta,
                    u16* Cb, int Mreal, int Nreal, int gridN,
                    const float* bias, int biasMode, int headPad) {
        dim3 grid(gridN, NPAD / 128);
        gemm_bf16_nt<<<grid, 256, 0, stream>>>(A, NPAD, kOffA, Bt, NPAD, kOffB,
                                               Kt, alpha, Cf, Cadd, beta, Cb,
                                               NPAD, Mreal, Nreal,
                                               bias, biasMode, headPad);
    };
    auto pack = [&](const float* src, int stride, int rows, int cols, u16* dst) {
        pack_bf16_kernel<<<egrid, 256, 0, stream>>>(src, stride, rows, cols, dst);
    };

    // ---- RWR (transposed recurrence): U_{s+1} = 0.3*U_s P + 0.7*E^T ----
    rowsum_kernel<<<NREAL, 256, 0, stream>>>(feature, rs);
    rwr_init_kernel<<<egrid, 256, 0, stream>>>(feature, rs, etas,
                                               PTb, Ub0, ETf, rg, Fb, Hf0);
    u16* Ua = Ub0; u16* Un = Ub1;
    for (int s = 0; s < 2; ++s) {
        gemm(Ua, 0, PTb, 0, NPAD, 0.3f, Uf, ETf, 0.7f, Un,
             NREAL, NREAL, 12, nullptr, 0, 0);
        rg_acc_kernel<<<egrid, 256, 0, stream>>>(rg, Uf, etas, s + 1);
        u16* t = Ua; Ua = Un; Un = t;
    }

    // ---- transformer layers ----
    u16*   HbCur = Fb;  float* HfCur = Hf0;
    u16*   HbOut[2] = {Hb1, Hb2};
    float* HfOut[2] = {Hf1, Hf0};
    const float rscale = 1.0f / sqrtf((float)DK);

    for (int l = 0; l < 2; ++l) {
        const float* wq = Wq + (size_t)l * D2;
        const float* wk = Wk + (size_t)l * D2;
        const float* wv = Wv + (size_t)l * D2;
        const float* wo = Wo + (size_t)l * D2;

        // Q = H Wq^T + bq  (head-padded bf16 columns)
        pack(wq, NREAL, NREAL, NREAL, Wb);
        gemm(HbCur, 0, Wb, 0, NPAD, 1.f, nullptr, nullptr, 0.f, Qb,
             NREAL, NREAL, 12, bq + (size_t)l * NREAL, 1, 1);
        // K = H Wk^T + bk
        pack(wk, NREAL, NREAL, NREAL, Wb);
        gemm(HbCur, 0, Wb, 0, NPAD, 1.f, nullptr, nullptr, 0.f, Kb,
             NREAL, NREAL, 12, bk + (size_t)l * NREAL, 1, 1);
        // V^T = Wv H^T + bv (row bias)  -> Vtb[d][N]
        pack(wv, NREAL, NREAL, NREAL, Wb);
        gemm(Wb, 0, HbCur, 0, NPAD, 1.f, nullptr, nullptr, 0.f, Vtb,
             NREAL, NREAL, 12, bv + (size_t)l * NREAL, 2, 0);

        for (int h = 0; h < 3; ++h) {
            // S_h = (Q_h K_h^T)/sqrt(dk)   (padded head K = 512, zeros in pads)
            gemm(Qb, h * DKP, Kb, h * DKP, DKP, rscale, Sfp, nullptr, 0.f,
                 nullptr, NREAL, NREAL, 12, nullptr, 0, 0);
            // attn_h = softmax(S_h * rg) -> bf16
            bias_softmax_bf16<<<NPAD, 256, 0, stream>>>(Sfp, rg, Sbb);
            // ctx_h = attn_h V_h = attn_h Vt_h^T -> CTXb columns [h*509, +509)
            gemm(Sbb, 0, Vtb + (size_t)h * DK * NPAD, 0, NPAD, 1.f,
                 nullptr, nullptr, 0.f, CTXb + (size_t)h * DK,
                 NREAL, DK, 4, nullptr, 0, 0);
        }
        zero_cols_kernel<<<(NPAD + 255) / 256, 256, 0, stream>>>(CTXb, NREAL);

        // H' = H + ctx Wo^T + bo
        pack(wo, NREAL, NREAL, NREAL, Wb);
        gemm(CTXb, 0, Wb, 0, NPAD, 1.f, HfOut[l], HfCur, 1.f, HbOut[l],
             NREAL, NREAL, 12, bo + (size_t)l * NREAL, 1, 0);
        HbCur = HbOut[l]; HfCur = HfOut[l];
    }

    // ---- gate: G = features @ gW[:, :d]^T + feature @ gW[:, d:]^T ----
    pack(gate_W, 2 * NREAL, NREAL, NREAL, Wb);
    gemm(HbCur, 0, Wb, 0, NPAD, 1.f, Gf, nullptr, 0.f, nullptr,
         NREAL, NREAL, 12, nullptr, 0, 0);
    pack(gate_W + NREAL, 2 * NREAL, NREAL, NREAL, Wb);
    gemm(Fb, 0, Wb, 0, NPAD, 1.f, Gf, Gf, 1.f, nullptr,
         NREAL, NREAL, 12, nullptr, 0, 0);

    gate_out_kernel<<<(int)((D2 + 255) / 256), 256, 0, stream>>>(
        Gf, gate_b, feature, faeture, out);
}
